// HypergraphConvolution_64811056496736
// MI455X (gfx1250) — compile-verified, matrix operand order checked by reading, not run on hardware
//
#include <hip/hip_runtime.h>

// Problem constants (from reference)
#define BATCH 8
#define NN    4096
#define EE    1024
#define DIN   256
#define DOUT  256
#define EPSV  1e-8f
#define DE_CHUNKS 16

typedef __attribute__((ext_vector_type(16))) __bf16 v16bf;
typedef __attribute__((ext_vector_type(8)))  float  v8f;

// ---------------------------------------------------------------------------
// Pack two fp32 into packed bf16 (native conversion; RNE)
// ---------------------------------------------------------------------------
__device__ __forceinline__ unsigned pack_bf16(float a, float b) {
    union { __bf16 h[2]; unsigned u; } p;
    p.h[0] = (__bf16)a;
    p.h[1] = (__bf16)b;
    return p.u;
}

union FragU {
    uint4  u[2];
    v16bf  v;
};

// Load one 16x32 bf16 A/B fragment for lane `lane` from an LDS tile row.
// ISA 7.12.2 (wave32): lanes 0-15 hold K=0..7 (v0-3) and K=16..23 (v4-7);
// lanes 16-31 hold K=8..15 and K=24..31.  kbase = (lane>=16)*8.
__device__ __forceinline__ v16bf load_frag(const unsigned short* rowbase, int kbase) {
    FragU f;
    f.u[0] = *(const uint4*)(rowbase + kbase);        // K = kbase .. kbase+7
    f.u[1] = *(const uint4*)(rowbase + kbase + 16);   // K = kbase+16 .. kbase+23
    return f.v;
}

// ---------------------------------------------------------------------------
// d_v^{-1/2}: one block per (b, n) row of H; 256-thread reduction over E
// ---------------------------------------------------------------------------
__global__ void hgc_dv_kernel(const float* __restrict__ H, float* __restrict__ dvs) {
    const int row = blockIdx.x;                 // flat b*N + n
    const float* hrow = H + (size_t)row * EE;
    float s = 0.f;
    for (int e = threadIdx.x; e < EE; e += 256) s += hrow[e];
    __shared__ float red[256];
    red[threadIdx.x] = s;
    __syncthreads();
    for (int off = 128; off > 0; off >>= 1) {
        if (threadIdx.x < off) red[threadIdx.x] += red[threadIdx.x + off];
        __syncthreads();
    }
    if (threadIdx.x == 0) {
        float d = red[0];
        d = d < EPSV ? EPSV : d;
        dvs[row] = rsqrtf(d);
    }
}

// ---------------------------------------------------------------------------
// d_e partial column sums (deterministic: fixed chunking, no float atomics)
// grid = (E/256, DE_CHUNKS, B)
// ---------------------------------------------------------------------------
__global__ void hgc_de_partial_kernel(const float* __restrict__ H, float* __restrict__ part) {
    const int b     = blockIdx.z;
    const int chunk = blockIdx.y;
    const int e     = blockIdx.x * 256 + threadIdx.x;
    const int nper  = NN / DE_CHUNKS;
    const float* p  = H + (size_t)b * NN * EE + (size_t)(chunk * nper) * EE + e;
    float s = 0.f;
    for (int n = 0; n < nper; ++n) {
        s += *p;
        p += EE;
    }
    part[((size_t)b * DE_CHUNKS + chunk) * EE + e] = s;
}

__global__ void hgc_de_final_kernel(const float* __restrict__ part, float* __restrict__ dei) {
    const int idx = blockIdx.x * 256 + threadIdx.x;  // flat b*E + e
    const int b = idx / EE;
    const int e = idx % EE;
    const float* p = part + (size_t)b * DE_CHUNKS * EE + e;
    float s = 0.f;
    for (int c = 0; c < DE_CHUNKS; ++c) {
        s += *p;
        p += EE;
    }
    s = s < EPSV ? EPSV : s;
    dei[idx] = 1.0f / s;
}

// ---------------------------------------------------------------------------
// Batched WMMA GEMM: C[b] = diag(rowscale[b]) * (op(A[b]) x B[b]) (+ bias)
//   fp32 global operands, converted to packed bf16 while staging into LDS,
//   fp32 accumulation via v_wmma_f32_16x16x32_bf16.
// Block tile 128x128, BK=32, 256 threads = 8 waves, wave tile 32x64 (2x4 wmma)
// TRANSA=1: A element (m,k) read as A[k*lda + m] (used for H^T); global loads
//   stay coalesced along the contiguous axis in both modes.
// All hot-loop addressing is pointer+constant-stride (no 64-bit muls in loop).
// ---------------------------------------------------------------------------
#define BM 128
#define BN 128
#define BK 32
#define LDS_PITCH 40   // 32 + 8 pad (ushorts); row stride 80B keeps 16B align

template <int TRANSA>
__global__ __launch_bounds__(256)
void hgc_wmma_gemm_kernel(const float* __restrict__ Ag, const float* __restrict__ Bg,
                          float* __restrict__ Cg,
                          const float* __restrict__ rowscale, const float* __restrict__ bias,
                          int K, int lda, int ldb, int ldc,
                          long long strideA, long long strideB, long long strideC,
                          long long strideScale) {
    __shared__ __align__(16) unsigned short As[BM][LDS_PITCH];
    __shared__ __align__(16) unsigned short Bs[BN][LDS_PITCH];

    const int tid  = threadIdx.x;
    const int lane = tid & 31;
    const int wave = tid >> 5;           // 0..7
    const int mwave = (wave >> 1) * 32;  // 4 waves along M
    const int nwave = (wave & 1) * 64;   // 2 waves along N
    const int kbase = (lane >> 4) * 8;   // fragment K base per ISA layout
    const int lrow  = lane & 15;

    const int m0 = blockIdx.y * BM;
    const int n0 = blockIdx.x * BN;
    const int bz = blockIdx.z;

    const float* A  = Ag + (size_t)bz * strideA;
    const float* Bm = Bg + (size_t)bz * strideB;
    float*       C  = Cg + (size_t)bz * strideC;
    const float* rs = rowscale + (size_t)bz * strideScale;

    // --- hoisted stage-in pointers (advance by constant strides per k-step) ---
    // A, non-trans: pairs along k. kp = even k, mr = row; 8 passes of 16 rows.
    const int kp   = (tid & 15) * 2;
    const int mrA  = tid >> 4;                 // 0..15
    // A, trans / B: coalesced along contiguous axis (m or n), pairs of k rows.
    const int mmA  = tid & 127;
    const int half = tid >> 7;                 // 0..1 -> k half (16 k each)

    const float* apN = A + (size_t)(m0 + mrA) * lda + kp;            // !TRANSA
    const float* apT = A + (size_t)(half * 16) * lda + (m0 + mmA);   //  TRANSA
    const float* bp0 = Bm + (size_t)(half * 16) * ldb + (n0 + mmA);

    unsigned short* asN = &As[mrA][kp];
    unsigned short* asT = &As[mmA][half * 16];
    unsigned short* bss = &Bs[mmA][half * 16];

    const int aStepN = BK;            // elements per k-step for apN
    const int aStepT = BK * lda;      // k-rows advance for apT
    const int bStep  = BK * ldb;

    v8f acc[2][4];
#pragma unroll
    for (int i = 0; i < 2; ++i)
#pragma unroll
        for (int j = 0; j < 4; ++j)
            acc[i][j] = {};

    for (int k0 = 0; k0 < K; k0 += BK) {
        // ---- stage A tile (BM x BK) into LDS as packed bf16 ----
        if (!TRANSA) {
            const float* p = apN;
            unsigned short* q = asN;
#pragma unroll
            for (int i = 0; i < 8; ++i) {           // 16 rows / pass
                const float2 v = *(const float2*)p;
                *(unsigned*)q = pack_bf16(v.x, v.y);
                p += (size_t)16 * lda;
                q += 16 * LDS_PITCH;
            }
            apN += aStepN;
        } else {
            const float* p = apT;
#pragma unroll
            for (int i = 0; i < 8; ++i) {           // 2 k-rows / pass
                const float f0 = p[0];
                const float f1 = p[lda];
                *(unsigned*)(asT + i * 2) = pack_bf16(f0, f1);
                p += (size_t)2 * lda;
            }
            apT += aStepT;
        }
        // ---- stage B tile (BK x BN) into LDS transposed: Bs[n][k] ----
        {
            const float* p = bp0;
#pragma unroll
            for (int i = 0; i < 8; ++i) {
                const float f0 = p[0];
                const float f1 = p[ldb];
                *(unsigned*)(bss + i * 2) = pack_bf16(f0, f1);
                p += (size_t)2 * ldb;
            }
            bp0 += bStep;
        }
        __syncthreads();

        v16bf afr[2];
        v16bf bfr[4];
#pragma unroll
        for (int mf = 0; mf < 2; ++mf)
            afr[mf] = load_frag(&As[mwave + mf * 16 + lrow][0], kbase);
#pragma unroll
        for (int nf = 0; nf < 4; ++nf)
            bfr[nf] = load_frag(&Bs[nwave + nf * 16 + lrow][0], kbase);

#pragma unroll
        for (int mf = 0; mf < 2; ++mf)
#pragma unroll
            for (int nf = 0; nf < 4; ++nf)
                acc[mf][nf] = __builtin_amdgcn_wmma_f32_16x16x32_bf16(
                    false, afr[mf], false, bfr[nf], (short)0, acc[mf][nf], false, false);

        __syncthreads();
    }

    // ---- epilogue: D layout (ISA 7.12.2): vgpr j -> M = j + 8*(lane>=16),
    //      N = lane&15.  Apply rowscale (+bias), store fp32 with strided ptr. ----
#pragma unroll
    for (int mf = 0; mf < 2; ++mf) {
        const int mbase = m0 + mwave + mf * 16 + (lane >> 4) * 8;
        const float* rp = rs + mbase;
#pragma unroll
        for (int nf = 0; nf < 4; ++nf) {
            const int col = n0 + nwave + nf * 16 + lrow;
            const float bcol = bias ? bias[col] : 0.0f;
            float* cp = C + (size_t)mbase * ldc + col;
            v8f a = acc[mf][nf];
#pragma unroll
            for (int j = 0; j < 8; ++j) {
                *cp = a[j] * rp[j] + bcol;
                cp += ldc;
            }
        }
    }
}

// ---------------------------------------------------------------------------
// Launcher:
//  out = diag(dv^-1/2) H diag(de^-1) H^T diag(dv^-1/2) (X W) + bias
// ---------------------------------------------------------------------------
extern "C" void kernel_launch(void* const* d_in, const int* in_sizes, int n_in,
                              void* d_out, int out_size, void* d_ws, size_t ws_size,
                              hipStream_t stream) {
    const float* x    = (const float*)d_in[0];   // [B,N,DIN]
    const float* H    = (const float*)d_in[1];   // [B,N,E]
    const float* W    = (const float*)d_in[2];   // [DIN,DOUT]
    const float* bias = (const float*)d_in[3];   // [DOUT]
    float* out = (float*)d_out;                  // [B,N,DOUT]

    float* ws = (float*)d_ws;
    size_t off = 0;
    float* dvs  = ws + off; off += (size_t)BATCH * NN;               // d_v^{-1/2}
    float* dei  = ws + off; off += (size_t)BATCH * EE;               // d_e^{-1}
    float* part = ws + off; off += (size_t)BATCH * DE_CHUNKS * EE;   // col-sum partials
    float* S    = ws + off; off += (size_t)BATCH * NN * DOUT;        // rowscaled X*W
    float* T    = ws + off; off += (size_t)BATCH * EE * DOUT;        // rowscaled H^T*S

    // degrees
    hgc_dv_kernel<<<BATCH * NN, 256, 0, stream>>>(H, dvs);
    hgc_de_partial_kernel<<<dim3(EE / 256, DE_CHUNKS, BATCH), 256, 0, stream>>>(H, part);
    hgc_de_final_kernel<<<(BATCH * EE) / 256, 256, 0, stream>>>(part, dei);

    // GEMM0: S = diag(dvs) * (X[b] (4096x256) x W (256x256))
    hgc_wmma_gemm_kernel<0><<<dim3(DOUT / BN, NN / BM, BATCH), 256, 0, stream>>>(
        x, W, S, dvs, nullptr,
        /*K*/ DIN, /*lda*/ DIN, /*ldb*/ DOUT, /*ldc*/ DOUT,
        (long long)NN * DIN, 0LL, (long long)NN * DOUT, (long long)NN);

    // GEMM1: T = diag(dei) * (H[b]^T (1024x4096) x S (4096x256))
    hgc_wmma_gemm_kernel<1><<<dim3(DOUT / BN, EE / BM, BATCH), 256, 0, stream>>>(
        H, S, T, dei, nullptr,
        /*K*/ NN, /*lda*/ EE, /*ldb*/ DOUT, /*ldc*/ DOUT,
        (long long)NN * EE, (long long)NN * DOUT, (long long)EE * DOUT, (long long)EE);

    // GEMM2: out = diag(dvs) * (H[b] (4096x1024) x T (1024x256)) + bias
    hgc_wmma_gemm_kernel<0><<<dim3(DOUT / BN, NN / BM, BATCH), 256, 0, stream>>>(
        H, T, out, dvs, bias,
        /*K*/ EE, /*lda*/ EE, /*ldb*/ DOUT, /*ldc*/ DOUT,
        (long long)NN * EE, (long long)EE * DOUT, (long long)NN * DOUT, (long long)NN);
}
